// LSTMEncoder_73624329388491
// MI455X (gfx1250) — compile-verified, matrix operand order checked by reading, not run on hardware
//
#include <hip/hip_runtime.h>
#include <hip/hip_bf16.h>

// ---------------------------------------------------------------------------
// LSTM encoder for MI455X (gfx1250, wave32, WMMA bf16 16x16x32).
//   B=256, T=128, E=300 (padded to 320), H=2048, 4H=8192.
// Per timestep: gates = h @ W_hh^T + x_t @ W_ih^T + (b_ih+b_hh), fused cell
// update, h double-buffered across step-kernel launches.
// ---------------------------------------------------------------------------

typedef __attribute__((ext_vector_type(16))) __bf16 v16bf;
typedef __attribute__((ext_vector_type(8)))  __bf16 v8bf;
typedef __attribute__((ext_vector_type(8)))  float  v8f;

#define LSTM_B   256
#define LSTM_T   128
#define LSTM_E   300
#define LSTM_EP  320      // E padded to multiple of 32
#define LSTM_H   2048
#define LSTM_4H  8192

union Frag16 {
    v16bf v;
    struct { v8bf lo, hi; } h;
};

__device__ __forceinline__ float fast_sigmoid(float x) {
    return 1.0f / (1.0f + __expf(-x));
}
__device__ __forceinline__ float fast_tanh(float x) {
    // tanh(x) = 1 - 2/(exp(2x)+1)
    return 1.0f - 2.0f / (__expf(2.0f * x) + 1.0f);
}

// f32 -> bf16 with optional column padding (zeros at col >= srcCols).
__global__ void cvt_pad_bf16(const float* __restrict__ src, __bf16* __restrict__ dst,
                             int rows, int srcCols, int dstCols) {
    long long n = (long long)rows * dstCols;
    for (long long i = (long long)blockIdx.x * blockDim.x + threadIdx.x;
         i < n; i += (long long)gridDim.x * blockDim.x) {
        int col = (int)(i % dstCols);
        int row = (int)(i / dstCols);
        float v = (col < srcCols) ? src[(long long)row * srcCols + col] : 0.0f;
        dst[i] = (__bf16)v;
    }
}

// Zero c (f32) and the initial h buffer (bf16).
__global__ void init_state(float* __restrict__ c, __bf16* __restrict__ h, int n) {
    for (int i = blockIdx.x * blockDim.x + threadIdx.x; i < n;
         i += gridDim.x * blockDim.x) {
        c[i] = 0.0f;
        h[i] = (__bf16)0.0f;
    }
}

// One LSTM timestep. Wave -> 32 batch rows x 16 hidden cols across all 4 gates.
// Grid: 128 blocks x 256 threads (8 waves) = 1024 waves covering
//       (256/32) m-tiles x (2048/16) n-tiles.
__global__ __launch_bounds__(256)
void lstm_step(const __bf16* __restrict__ hin,   // [B][H] bf16
               __bf16*       __restrict__ hout,  // [B][H] bf16
               float*        __restrict__ c,     // [B][H] f32 (in-place)
               const __bf16* __restrict__ xb,    // [B][T][EP] bf16
               const __bf16* __restrict__ Wih,   // [4H][EP] bf16
               const __bf16* __restrict__ Whh,   // [4H][H]  bf16
               const float*  __restrict__ b_ih,  // [4H] f32
               const float*  __restrict__ b_hh,  // [4H] f32
               int t,
               float* __restrict__ outp)         // [B][H] f32 or nullptr
{
    const int lane  = threadIdx.x & 31;
    const int wid   = threadIdx.x >> 5;
    const int gw    = blockIdx.x * 8 + wid;
    const int nTile = gw & 127;          // 0..127  -> 16 hidden cols
    const int mTile = gw >> 7;           // 0..7    -> 32 batch rows
    const int h0    = nTile * 16;
    const int m0    = mTile * 32;
    const int ln    = lane & 15;         // row/col within tile
    const int lg    = lane >> 4;         // K-half select (0 or 1)

    v8f acc[4][2];
#pragma unroll
    for (int g = 0; g < 4; ++g)
#pragma unroll
        for (int mi = 0; mi < 2; ++mi)
            acc[g][mi] = (v8f)(0.0f);

    // ---- recurrent part: h @ W_hh^T, K = 2048 ----
    {
        const __bf16* a0p = hin + (size_t)(m0 + ln) * LSTM_H + lg * 8;
        const __bf16* a1p = a0p + (size_t)16 * LSTM_H;
        const __bf16* bp[4];
#pragma unroll
        for (int g = 0; g < 4; ++g)
            bp[g] = Whh + (size_t)(g * LSTM_H + h0 + ln) * LSTM_H + lg * 8;

        for (int k = 0; k < LSTM_H; k += 32) {
            Frag16 a0, a1;
            a0.h.lo = *(const v8bf*)(a0p + k);
            a0.h.hi = *(const v8bf*)(a0p + k + 16);
            a1.h.lo = *(const v8bf*)(a1p + k);
            a1.h.hi = *(const v8bf*)(a1p + k + 16);
#pragma unroll
            for (int g = 0; g < 4; ++g) {
                Frag16 b;
                b.h.lo = *(const v8bf*)(bp[g] + k);
                b.h.hi = *(const v8bf*)(bp[g] + k + 16);
                acc[g][0] = __builtin_amdgcn_wmma_f32_16x16x32_bf16(
                    false, a0.v, false, b.v, (short)0, acc[g][0], false, false);
                acc[g][1] = __builtin_amdgcn_wmma_f32_16x16x32_bf16(
                    false, a1.v, false, b.v, (short)0, acc[g][1], false, false);
            }
        }
    }

    // ---- input part: x_t @ W_ih^T, K = 320 (padded) ----
    {
        const __bf16* a0p = xb + ((size_t)(m0 + ln) * LSTM_T + t) * LSTM_EP + lg * 8;
        const __bf16* a1p = a0p + (size_t)16 * LSTM_T * LSTM_EP;
        const __bf16* bp[4];
#pragma unroll
        for (int g = 0; g < 4; ++g)
            bp[g] = Wih + (size_t)(g * LSTM_H + h0 + ln) * LSTM_EP + lg * 8;

        for (int k = 0; k < LSTM_EP; k += 32) {
            Frag16 a0, a1;
            a0.h.lo = *(const v8bf*)(a0p + k);
            a0.h.hi = *(const v8bf*)(a0p + k + 16);
            a1.h.lo = *(const v8bf*)(a1p + k);
            a1.h.hi = *(const v8bf*)(a1p + k + 16);
#pragma unroll
            for (int g = 0; g < 4; ++g) {
                Frag16 b;
                b.h.lo = *(const v8bf*)(bp[g] + k);
                b.h.hi = *(const v8bf*)(bp[g] + k + 16);
                acc[g][0] = __builtin_amdgcn_wmma_f32_16x16x32_bf16(
                    false, a0.v, false, b.v, (short)0, acc[g][0], false, false);
                acc[g][1] = __builtin_amdgcn_wmma_f32_16x16x32_bf16(
                    false, a1.v, false, b.v, (short)0, acc[g][1], false, false);
            }
        }
    }

    // ---- epilogue: bias + activations + cell update on C-fragment layout ----
    // C layout: VGPR r <-> M = r + 8*lg, N = lane&15.
    const int hcol = h0 + ln;
    float bias[4];
#pragma unroll
    for (int g = 0; g < 4; ++g)
        bias[g] = b_ih[g * LSTM_H + hcol] + b_hh[g * LSTM_H + hcol];

#pragma unroll
    for (int mi = 0; mi < 2; ++mi) {
#pragma unroll
        for (int r = 0; r < 8; ++r) {
            const int brow = m0 + mi * 16 + lg * 8 + r;
            const size_t idx = (size_t)brow * LSTM_H + hcol;
            float iv = fast_sigmoid(acc[0][mi][r] + bias[0]);
            float fv = fast_sigmoid(acc[1][mi][r] + bias[1]);
            float gv = fast_tanh   (acc[2][mi][r] + bias[2]);
            float ov = fast_sigmoid(acc[3][mi][r] + bias[3]);
            float cn = fv * c[idx] + iv * gv;
            c[idx] = cn;
            float hn = ov * fast_tanh(cn);
            hout[idx] = (__bf16)hn;
            if (outp) outp[idx] = hn;
        }
    }
}

extern "C" void kernel_launch(void* const* d_in, const int* in_sizes, int n_in,
                              void* d_out, int out_size, void* d_ws, size_t ws_size,
                              hipStream_t stream) {
    const float* x    = (const float*)d_in[0];   // [B][T][E]
    const float* W_ih = (const float*)d_in[1];   // [4H][E]
    const float* W_hh = (const float*)d_in[2];   // [4H][H]
    const float* b_ih = (const float*)d_in[3];   // [4H]
    const float* b_hh = (const float*)d_in[4];   // [4H]
    float* out = (float*)d_out;                  // [B][H]

    // Workspace layout (all offsets 256B-aligned).
    char* ws = (char*)d_ws;
    __bf16* xb   = (__bf16*)ws;                       // B*T*EP bf16 = 20 MB
    ws += (size_t)LSTM_B * LSTM_T * LSTM_EP * 2;
    __bf16* wihb = (__bf16*)ws;                       // 4H*EP bf16  = 5 MB
    ws += (size_t)LSTM_4H * LSTM_EP * 2;
    __bf16* whhb = (__bf16*)ws;                       // 4H*H bf16   = 32 MB
    ws += (size_t)LSTM_4H * LSTM_H * 2;
    __bf16* hping = (__bf16*)ws;                      // B*H bf16    = 1 MB
    ws += (size_t)LSTM_B * LSTM_H * 2;
    __bf16* hpong = (__bf16*)ws;                      // B*H bf16    = 1 MB
    ws += (size_t)LSTM_B * LSTM_H * 2;
    float* cbuf = (float*)ws;                         // B*H f32     = 2 MB

    // 1) bf16 conversions (W_hh 32MB becomes L2-resident for all 128 steps).
    {
        long long n;
        n = (long long)LSTM_B * LSTM_T * LSTM_EP;
        cvt_pad_bf16<<<(int)((n + 255) / 256), 256, 0, stream>>>(
            x, xb, LSTM_B * LSTM_T, LSTM_E, LSTM_EP);
        n = (long long)LSTM_4H * LSTM_EP;
        cvt_pad_bf16<<<(int)((n + 255) / 256), 256, 0, stream>>>(
            W_ih, wihb, LSTM_4H, LSTM_E, LSTM_EP);
        n = (long long)LSTM_4H * LSTM_H;
        cvt_pad_bf16<<<(int)((n + 255) / 256), 256, 0, stream>>>(
            W_hh, whhb, LSTM_4H, LSTM_H, LSTM_H);
    }

    // 2) zero c and the t=0 h buffer (re-done every call -> deterministic).
    {
        int n = LSTM_B * LSTM_H;
        init_state<<<(n + 255) / 256, 256, 0, stream>>>(cbuf, hping, n);
    }

    // 3) 128 sequential timesteps; kernel boundary = inter-step barrier.
    for (int t = 0; t < LSTM_T; ++t) {
        const __bf16* hi = (t & 1) ? hpong : hping;
        __bf16*       ho = (t & 1) ? hping : hpong;
        lstm_step<<<128, 256, 0, stream>>>(
            hi, ho, cbuf, xb, wihb, whhb, b_ih, b_hh, t,
            (t == LSTM_T - 1) ? out : nullptr);
    }
}
